// SATSolver_29643864277124
// MI455X (gfx1250) — compile-verified, matrix operand order checked by reading, not run on hardware
//
#include <hip/hip_runtime.h>
#include <hip/hip_bf16.h>

// Problem constants (match reference)
#define V_N 30000
#define C_N 126000
#define E_N 378000
#define G_N 64
#define ROUNDS_N 4

typedef _Float16 f16;
typedef __attribute__((ext_vector_type(16))) _Float16 v16h;
typedef __attribute__((ext_vector_type(8)))  float    v8f;

// ---------------------------------------------------------------- helpers
__device__ __forceinline__ float softplusf(float x) {
    float ax = fabsf(x);
    return log1pf(__expf(-ax)) + fmaxf(x, 0.0f);
}
__device__ __forceinline__ float sigm(float x) { return 1.0f / (1.0f + __expf(-x)); }

__device__ __forceinline__ unsigned hashu(unsigned x) {
    x ^= x >> 16; x *= 0x7feb352dU;
    x ^= x >> 15; x *= 0x846ca68bU;
    x ^= x >> 16; return x;
}
__device__ __forceinline__ unsigned mix3(unsigned a, unsigned b, unsigned c) {
    return hashu(a * 0x9E3779B9U ^ b * 0x85EBCA6BU ^ c * 0xC2B2AE35U);
}
__device__ __forceinline__ float u01(unsigned s) {
    return (float)(hashu(s) >> 8) * (1.0f / 16777216.0f) + 1e-7f;
}
__device__ __forceinline__ float gaussn(unsigned s) {
    float a = u01(s), b = u01(s ^ 0x68E31DA4U);
    return sqrtf(-2.0f * logf(a)) * __cosf(6.28318530718f * b);
}

// ---------------------------------------------------------------- WMMA GEMM
// Out[M,N] = act(A[M,K](f32) x W[K,N] + bias[N]), W pre-packed to f16 fragments.
// K must be a multiple of 32 (callers zero-pad). N = NT*16.
// Block = 256 threads = 8 waves; each wave owns a 16-row strip and all NT col
// tiles, reusing one A fragment for NT v_wmma ops per K-step. No LDS, no
// barriers: A path is 4x global_load_b128 + cvt per K-step, B path is one
// 32-byte contiguous load per tile (L2-resident pre-packed weights).
template <int NT>
__global__ __launch_bounds__(256) void gemm_wmma(
    const float* __restrict__ A, const f16* __restrict__ Bp,
    const float* __restrict__ bias, float* __restrict__ Out,
    int M, int K, float slope)
{
    const int tid  = threadIdx.x;
    const int wave = tid >> 5;
    const int lane = tid & 31;
    const int half = lane >> 4;
    const int rowA = blockIdx.x * 128 + (wave << 4) + (lane & 15);
    const int rowC = (rowA < M) ? rowA : (M - 1);               // clamp loads
    const float* __restrict__ Arow = A + (size_t)rowC * K + (half << 3);

    v8f zero = {};
    v8f acc[NT];
    #pragma unroll
    for (int n = 0; n < NT; ++n) acc[n] = zero;

    const int KT = K >> 5;
    for (int kt = 0; kt < KT; ++kt) {
        const float4* __restrict__ ap = (const float4*)(Arow + (kt << 5));
        float4 s0 = ap[0];   // k_rel 0..3
        float4 s1 = ap[1];   // k_rel 4..7
        float4 s2 = ap[4];   // k_rel 16..19
        float4 s3 = ap[5];   // k_rel 20..23
        if (kt + 1 < KT)
            __builtin_prefetch(Arow + ((kt + 1) << 5), 0, 1);   // global_prefetch_b8
        v16h a;
        a[0] = (f16)s0.x;  a[1] = (f16)s0.y;  a[2]  = (f16)s0.z;  a[3]  = (f16)s0.w;
        a[4] = (f16)s1.x;  a[5] = (f16)s1.y;  a[6]  = (f16)s1.z;  a[7]  = (f16)s1.w;
        a[8] = (f16)s2.x;  a[9] = (f16)s2.y;  a[10] = (f16)s2.z;  a[11] = (f16)s2.w;
        a[12] = (f16)s3.x; a[13] = (f16)s3.y; a[14] = (f16)s3.z;  a[15] = (f16)s3.w;

        const f16* __restrict__ bbase = Bp + (((size_t)kt * NT) << 9) + (lane << 4);
        #pragma unroll
        for (int n = 0; n < NT; ++n) {
            v16h b = *(const v16h*)(bbase + ((size_t)n << 9));
            acc[n] = __builtin_amdgcn_wmma_f32_16x16x32_f16(false, a, false, b,
                                                            (short)0, acc[n], false, false);
        }
    }

    // D layout: row = r + 8*(lane>>4), col = lane&15 (per 16x16 tile)
    const int N    = NT * 16;
    const int col0 = lane & 15;
    const int rb   = blockIdx.x * 128 + (wave << 4) + (half << 3);
    #pragma unroll
    for (int n = 0; n < NT; ++n) {
        const int col = n * 16 + col0;
        const float bv = bias[col];
        #pragma unroll
        for (int r = 0; r < 8; ++r) {
            const int row = rb + r;
            if (row < M) {
                float x = acc[n][r] + bv;
                if (slope < 1.0f) x = (x > 0.0f) ? x : slope * x;
                Out[(size_t)row * N + col] = x;
            }
        }
    }
}

// Pack W[Ka,N] (f32 row-major) into per-(ktile,ntile,lane) fragment-ordered f16,
// zero-padded to Kp (multiple of 32).
__global__ void k_pack_wfrag(const float* __restrict__ W, f16* __restrict__ Bp,
                             int Ka, int N, int Kp) {
    int idx = blockIdx.x * 256 + threadIdx.x;
    int NT = N >> 4;
    int total = (Kp >> 5) * NT * 512;
    if (idx >= total) return;
    int t    = idx & 511;
    int tile = idx >> 9;
    int kt   = tile / NT;
    int nt   = tile - kt * NT;
    int lane = t >> 4;
    int i    = t & 15;
    int col  = nt * 16 + (lane & 15);
    int kh   = lane >> 4;
    int j    = i >> 1, e = i & 1;
    int krel = ((j >> 2) << 4) + (kh << 3) + ((j & 3) << 1) + e;
    int k    = (kt << 5) + krel;
    float v  = (k < Ka) ? W[(size_t)k * N + col] : 0.0f;
    Bp[idx]  = (f16)v;
}

// ---------------------------------------------------------------- small kernels
__global__ void k_fill(float* p, float v, size_t n) {
    size_t i = (size_t)blockIdx.x * 256 + threadIdx.x;
    if (i < n) p[i] = v;
}
__global__ void k_init_flags(int* f) { if (threadIdx.x == 0) { f[0] = 0; f[1] = 0; } }
__global__ void k_set_int(int* p, int v) { if (threadIdx.x == 0) *p = v; }

__global__ void k_edge_degree(const int* __restrict__ lit, float* deg, int E) {
    int i = blockIdx.x * 256 + threadIdx.x;
    if (i < E) atomicAdd(&deg[lit[i]], 1.0f);
}
__global__ void k_deg_weights(const float* __restrict__ deg, float* dw, float* vdw, int V) {
    int i = blockIdx.x * 256 + threadIdx.x;
    if (i < 2 * V) dw[i] = rsqrtf(fmaxf(deg[i], 1.0f));
    if (i < V)     vdw[i] = 4.0f * rsqrtf(fmaxf(deg[i] + deg[i + V], 1.0f));
}
__global__ void k_count_gid(const int* __restrict__ gid, float* cnt, int n) {
    int i = blockIdx.x * 256 + threadIdx.x;
    if (i < n) atomicAdd(&cnt[gid[i]], 1.0f);
}
__global__ void k_inv(const float* cnt, float* inv, int G) {
    int i = blockIdx.x * 256 + threadIdx.x;
    if (i < G) inv[i] = (cnt[i] > 0.0f) ? 1.0f / cnt[i] : 0.0f;
}

// concat [variables(64) | noise(4) | pad->96] -> A[V,96]
__global__ void k_assemble_vq(const float* __restrict__ vars, float* A, int V, int rnd) {
    size_t idx = (size_t)blockIdx.x * 256 + threadIdx.x;
    if (idx >= (size_t)V * 96) return;
    int i = (int)(idx / 96), c = (int)(idx % 96);
    float v = 0.0f;
    if (c < 64)      v = vars[(size_t)i * 64 + c];
    else if (c < 68) v = gaussn(mix3(rnd * 3 + 0, i, c));
    A[idx] = v;
}
// concat [clause_state(64) | mask(1) | noise(4) | pad->96] -> A[C,96]
__global__ void k_assemble_cq(const float* __restrict__ cs, const float* __restrict__ mask,
                              float* A, int C, int rnd) {
    size_t idx = (size_t)blockIdx.x * 256 + threadIdx.x;
    if (idx >= (size_t)C * 96) return;
    int i = (int)(idx / 96), c = (int)(idx % 96);
    float v = 0.0f;
    if (c < 64)       v = cs[(size_t)i * 64 + c];
    else if (c == 64) v = mask[i];
    else if (c < 69)  v = gaussn(mix3(rnd * 3 + 1, i, c));
    A[idx] = v;
}

// t[c,q] += softplus(+-var_query)  over edges
__global__ void k_spmm_cl_soft(const float* __restrict__ vq, const int* __restrict__ lit,
                               const int* __restrict__ cl, float* t, int E, int V) {
    size_t idx = (size_t)blockIdx.x * 256 + threadIdx.x;
    if (idx >= (size_t)E * 64) return;
    int e = (int)(idx >> 6), q = (int)(idx & 63);
    int l = lit[e];
    float x = (l < V) ? vq[(size_t)l * 64 + q] : -vq[(size_t)(l - V) * 64 + q];
    atomicAdd(&t[(size_t)cl[e] * 64 + q], softplusf(x));
}
// cval = exp(-t)*cq ; t <- cgrad = exp(-t)
__global__ void k_cval(const float* __restrict__ cq, float* t, float* cval, size_t n) {
    size_t i = (size_t)blockIdx.x * 256 + threadIdx.x;
    if (i >= n) return;
    float e = __expf(-t[i]);
    cval[i] = e * cq[i];
    t[i] = e;
}
// gback[l,q] += cval[c,q] over edges
__global__ void k_gback(const float* __restrict__ cval, const int* __restrict__ lit,
                        const int* __restrict__ cl, float* gb, int E) {
    size_t idx = (size_t)blockIdx.x * 256 + threadIdx.x;
    if (idx >= (size_t)E * 64) return;
    int e = (int)(idx >> 6), q = (int)(idx & 63);
    atomicAdd(&gb[(size_t)lit[e] * 64 + q], cval[(size_t)cl[e] * 64 + q]);
}
// variables_grad = (d clause_loss / d var_query) * var_degree_weight
__global__ void k_vgrad(const float* __restrict__ vq, const float* __restrict__ gb,
                        const float* __restrict__ vdw, float* vg, int V) {
    size_t idx = (size_t)blockIdx.x * 256 + threadIdx.x;
    if (idx >= (size_t)V * 64) return;
    int i = (int)(idx >> 6), q = (int)(idx & 63);
    float s = sigm(vq[idx]);
    float g = -s * gb[idx] + (1.0f - s) * gb[(size_t)(i + V) * 64 + q];
    vg[idx] = g * vdw[i];
}
// clause_unit = [state | cval*4 | cgrad] * mask  -> A[C,192]
__global__ void k_assemble_cu(const float* __restrict__ cs, const float* __restrict__ cval,
                              const float* __restrict__ cgrad, const float* __restrict__ mask,
                              float* A, int C) {
    size_t idx = (size_t)blockIdx.x * 256 + threadIdx.x;
    if (idx >= (size_t)C * 192) return;
    int i = (int)(idx / 192), c = (int)(idx % 192);
    float v;
    if (c < 64)       v = cs[(size_t)i * 64 + c];
    else if (c < 128) v = cval[(size_t)i * 64 + (c - 64)] * 4.0f;
    else              v = cgrad[(size_t)i * 64 + (c - 128)];
    A[idx] = v * mask[i];
}
// vl[l,q] += clause_data[c, q] (first 64 cols, stride 128)
__global__ void k_spmm_adj(const float* __restrict__ cdata, const int* __restrict__ lit,
                           const int* __restrict__ cl, float* vl, int E) {
    size_t idx = (size_t)blockIdx.x * 256 + threadIdx.x;
    if (idx >= (size_t)E * 64) return;
    int e = (int)(idx >> 6), q = (int)(idx & 63);
    atomicAdd(&vl[(size_t)lit[e] * 64 + q], cdata[(size_t)cl[e] * 128 + q]);
}
__global__ void k_scale_rows(float* vl, const float* __restrict__ dw, size_t n) {
    size_t i = (size_t)blockIdx.x * 256 + threadIdx.x;
    if (i < n) vl[i] *= dw[i >> 6];
}

// ---- PairNorm (3 passes); X has row stride `stride`, 64 cols
__global__ void k_pn_sum(const float* __restrict__ X, int stride, const int* __restrict__ gid,
                         float* sum, int n) {
    size_t idx = (size_t)blockIdx.x * 256 + threadIdx.x;
    if (idx >= (size_t)n * 64) return;
    int i = (int)(idx >> 6), q = (int)(idx & 63);
    atomicAdd(&sum[gid[i] * 64 + q], X[(size_t)i * stride + q]);
}
__global__ void k_pn_center(const float* __restrict__ X, int stride, const int* __restrict__ gid,
                            const float* __restrict__ sum, const float* __restrict__ inv,
                            float* XC, float* sq, int n) {
    size_t idx = (size_t)blockIdx.x * 256 + threadIdx.x;
    if (idx >= (size_t)n * 64) return;
    int i = (int)(idx >> 6), q = (int)(idx & 63);
    int g = gid[i];
    float xc = X[(size_t)i * stride + q] - sum[g * 64 + q] * inv[g];
    XC[idx] = xc;
    atomicAdd(&sq[g * 64 + q], xc * xc);
}
// XC = (XC * rsqrt(var+eps)) * 0.25 + 0.1*old   (in place)
__global__ void k_pn_finish(const int* __restrict__ gid, const float* __restrict__ sq,
                            const float* __restrict__ inv, const float* __restrict__ old,
                            float* XC, int n) {
    size_t idx = (size_t)blockIdx.x * 256 + threadIdx.x;
    if (idx >= (size_t)n * 64) return;
    int i = (int)(idx >> 6), q = (int)(idx & 63);
    int g = gid[i];
    float x = XC[idx] * rsqrtf(sq[g * 64 + q] * inv[g] + 1e-6f);
    XC[idx] = x * 0.25f + 0.1f * old[idx];
}

// unit = [vgrad | variables | vl_pos | vl_neg] -> A[V,256]
__global__ void k_assemble_unit(const float* __restrict__ vg, const float* __restrict__ vars,
                                const float* __restrict__ vl, float* A, int V) {
    size_t idx = (size_t)blockIdx.x * 256 + threadIdx.x;
    if (idx >= (size_t)V * 256) return;
    int i = (int)(idx >> 8), c = (int)(idx & 255);
    float v;
    if (c < 64)       v = vg[(size_t)i * 64 + c];
    else if (c < 128) v = vars[(size_t)i * 64 + (c - 64)];
    else if (c < 192) v = vl[(size_t)i * 64 + (c - 128)];
    else              v = vl[(size_t)(i + V_N) * 64 + (c - 192)];
    A[idx] = v;
}

// logits = hid @ w(64x1) + b + logistic noise
__global__ void k_logits(const float* __restrict__ hid, const float* __restrict__ w,
                         const float* __restrict__ b, float* lg, int V, int rnd) {
    int i = blockIdx.x * 256 + threadIdx.x;
    if (i >= V) return;
    float s = b[0];
    #pragma unroll 8
    for (int j = 0; j < 64; ++j) s += hid[(size_t)i * 64 + j] * w[j];
    const float eps = 1e-5f;
    float u  = u01(mix3(rnd * 3 + 2, i, 0x777));
    float sv = (1.0f - eps) - (1.0f - 2.0f * eps) * u;
    lg[i] = s + logf(sv / (1.0f - sv));
}

__global__ void k_eval_edges(const float* __restrict__ lg, const int* __restrict__ lit,
                             const int* __restrict__ cl, float* t2, float* satv, int E, int V) {
    int e = blockIdx.x * 256 + threadIdx.x;
    if (e >= E) return;
    int l = lit[e], c = cl[e];
    int v = (l < V) ? l : l - V;
    float x = (l < V) ? lg[v] : -lg[v];
    atomicAdd(&t2[c], softplusf(x));
    float a = roundf(sigm(lg[v]));
    atomicAdd(&satv[c], (l < V) ? a : 1.0f - a);
}
__global__ void k_eval_clauses(const float* __restrict__ t2, const float* __restrict__ satv,
                               const float* __restrict__ mask, const int* __restrict__ gid,
                               float* gloss, int* satflag, int C) {
    int c = blockIdx.x * 256 + threadIdx.x;
    if (c >= C) return;
    float cv = __expf(-t2[c]) * mask[c];
    atomicAdd(&gloss[gid[c]], cv * (-logf(1.0f - cv + 1e-10f)));
    if (satv[c] <= 0.0f) atomicAnd(satflag, 0);
}
__global__ void k_finish(const float* __restrict__ gloss, float* loss_accum,
                         int* flags, const int* satflag) {
    int g = threadIdx.x;
    int done = flags[0];
    if (!done && g < G_N) loss_accum[g] += sqrtf(gloss[g] + 1e-6f) - 0.001f;
    __syncthreads();
    if (g == 0 && !done) {
        flags[1] += 1;
        if (*satflag) flags[0] = 1;
    }
}
__global__ void k_commit(float* dst, const float* __restrict__ src,
                         const int* __restrict__ flags, size_t n) {
    size_t i = (size_t)blockIdx.x * 256 + threadIdx.x;
    if (i < n && flags[0] == 0) dst[i] = src[i];
}
__global__ void k_write_out(float* out, const float* __restrict__ ll,
                            const float* __restrict__ la, const int* __restrict__ flags, int V) {
    int i = blockIdx.x * 256 + threadIdx.x;
    if (i < V)              out[i] = ll[i];
    else if (i < V + G_N)   out[i] = la[i - V] * (1.0f / (float)ROUNDS_N);
    else if (i == V + G_N)  ((int*)out)[i] = flags[1];
}

// ---------------------------------------------------------------- host side
static inline dim3 gs(size_t n) { return dim3((unsigned)((n + 255) / 256)); }

static inline void gemm(hipStream_t st, const float* A, const f16* Bp, const float* b,
                        float* Out, int M, int K, int N, float slope) {
    dim3 grid((M + 127) / 128);
    if (N == 64) gemm_wmma<4><<<grid, 256, 0, st>>>(A, Bp, b, Out, M, K, slope);
    else         gemm_wmma<8><<<grid, 256, 0, st>>>(A, Bp, b, Out, M, K, slope);
}

extern "C" void kernel_launch(void* const* d_in, const int* in_sizes, int n_in,
                              void* d_out, int out_size, void* d_ws, size_t ws_size,
                              hipStream_t stream) {
    (void)in_sizes; (void)n_in; (void)out_size; (void)ws_size;
    const int V = V_N, C = C_N, E = E_N;

    // params pytree flattened: dict keys sorted, lists in order, {b,w} sorted
    const float* cm_b0 = (const float*)d_in[0];   // clause_mlp  L0 b [128]
    const float* cm_w0 = (const float*)d_in[1];   //             L0 w [192,128]
    const float* cm_b1 = (const float*)d_in[2];   //             L1 b [128]
    const float* cm_w1 = (const float*)d_in[3];   //             L1 w [128,128]
    const float* cq_b0 = (const float*)d_in[4];   // clauses_query L0 b [64]
    const float* cq_w0 = (const float*)d_in[5];   //               L0 w [69,64]
    const float* cq_b1 = (const float*)d_in[6];   //               L1 b [64]
    const float* cq_w1 = (const float*)d_in[7];   //               L1 w [64,64]
    const float* ug_b0 = (const float*)d_in[8];   // update_gate L0 b [128]
    const float* ug_w0 = (const float*)d_in[9];   //             L0 w [256,128]
    const float* ug_b1 = (const float*)d_in[10];  //             L1 b [128]
    const float* ug_w1 = (const float*)d_in[11];  //             L1 w [128,128]
    const float* ug_b2 = (const float*)d_in[12];  //             L2 b [64]
    const float* ug_w2 = (const float*)d_in[13];  //             L2 w [128,64]
    const float* vo_b0 = (const float*)d_in[14];  // variables_output L0 b [64]
    const float* vo_w0 = (const float*)d_in[15];  //                  L0 w [64,64]
    const float* vo_b1 = (const float*)d_in[16];  //                  L1 b [1]
    const float* vo_w1 = (const float*)d_in[17];  //                  L1 w [64,1]
    const float* vq_b0 = (const float*)d_in[18];  // variables_query L0 b [64]
    const float* vq_w0 = (const float*)d_in[19];  //                 L0 w [68,64]
    const float* vq_b1 = (const float*)d_in[20];  //                 L1 b [64]
    const float* vq_w1 = (const float*)d_in[21];  //                 L1 w [64,64]
    const float* mask      = (const float*)d_in[22];
    const int*   lit_idx   = (const int*)d_in[23];
    const int*   clause_idx= (const int*)d_in[24];
    const int*   cgid      = (const int*)d_in[25];
    const int*   vgid      = (const int*)d_in[26];

    // ---- workspace bump allocator (32-byte aligned allocations)
    float* ws = (float*)d_ws;
    size_t off = 0;
    auto FA = [&](size_t n) { float* r = ws + off; off += (n + 7) & ~(size_t)7; return r; };

    float* variables    = FA((size_t)V * 64);
    float* clause_state = FA((size_t)C * 64);
    float* last_logits  = FA(V);
    float* loss_accum   = FA(G_N);
    int*   flags        = (int*)FA(16);          // [0]=done [1]=step_count
    int*   satflag      = flags + 2;
    float* lit_degree   = FA((size_t)2 * V);
    float* dw           = FA((size_t)2 * V);
    float* vdw          = FA(V);
    float* ccnt = FA(G_N); float* vcnt = FA(G_N);
    float* inv_c = FA(G_N); float* inv_v = FA(G_N);
    float* pnsum = FA(G_N * 64); float* pnsq = FA(G_N * 64);

    // f16 fragment-packed weight region (padded K sizes)
    f16* hp = (f16*)FA(62000);
    auto HA = [&](size_t n) { f16* r = hp; hp += (n + 15) & ~(size_t)15; return r; };
    f16* wvq0p = HA(96 * 64);    // variables_query L0 (68 -> pad 96)
    f16* wvq1p = HA(64 * 64);
    f16* wcq0p = HA(96 * 64);    // clauses_query L0 (69 -> pad 96)
    f16* wcq1p = HA(64 * 64);
    f16* wcm0p = HA(192 * 128);
    f16* wcm1p = HA(128 * 128);
    f16* wug0p = HA(256 * 128);
    f16* wug1p = HA(128 * 128);
    f16* wug2p = HA(128 * 64);
    f16* wvo0p = HA(64 * 64);

    float* S_big   = FA((size_t)C * 192);   // concat buffers (cq / clause_unit / var unit)
    float* S_h1    = FA((size_t)C * 128);   // hidden
    float* S_cdata = FA((size_t)C * 128);   // clause_data
    float* cquery  = FA((size_t)C * 64);
    float* t_cgrad = FA((size_t)C * 64);    // t -> cgrad (in place)
    float* cval    = FA((size_t)C * 64);
    float* cand_c  = FA((size_t)C * 64);    // new clause state candidate
    float* t2      = FA(C);
    float* satv    = FA(C);
    float* gloss   = FA(G_N);
    float* vquery  = FA((size_t)V * 64);
    float* v_h1    = FA((size_t)V * 128);
    float* v_h2    = FA((size_t)V * 128);
    float* nv      = FA((size_t)V * 64);
    float* cand_v  = FA((size_t)V * 64);
    float* v_hid   = FA((size_t)V * 64);
    float* logitsb = FA(V);
    float* vgrad   = FA((size_t)V * 64);
    float* vl      = FA((size_t)2 * V * 64);
    float* gback   = FA((size_t)2 * V * 64);

    // ---- setup (re-done every call: deterministic)
    k_fill<<<gs((size_t)V * 64), 256, 0, stream>>>(variables, 1.0f, (size_t)V * 64);
    k_fill<<<gs((size_t)C * 64), 256, 0, stream>>>(clause_state, 1.0f, (size_t)C * 64);
    k_fill<<<gs(V), 256, 0, stream>>>(last_logits, 0.0f, V);
    k_fill<<<gs(G_N), 256, 0, stream>>>(loss_accum, 0.0f, G_N);
    k_init_flags<<<1, 1, 0, stream>>>(flags);
    k_fill<<<gs(2 * V), 256, 0, stream>>>(lit_degree, 0.0f, 2 * V);
    k_fill<<<gs(G_N), 256, 0, stream>>>(ccnt, 0.0f, G_N);
    k_fill<<<gs(G_N), 256, 0, stream>>>(vcnt, 0.0f, G_N);
    k_edge_degree<<<gs(E), 256, 0, stream>>>(lit_idx, lit_degree, E);
    k_deg_weights<<<gs(2 * V), 256, 0, stream>>>(lit_degree, dw, vdw, V);
    k_count_gid<<<gs(C), 256, 0, stream>>>(cgid, ccnt, C);
    k_count_gid<<<gs(V), 256, 0, stream>>>(vgid, vcnt, V);
    k_inv<<<gs(G_N), 256, 0, stream>>>(ccnt, inv_c, G_N);
    k_inv<<<gs(G_N), 256, 0, stream>>>(vcnt, inv_v, G_N);
    k_pack_wfrag<<<gs(96 * 64), 256, 0, stream>>>(vq_w0, wvq0p, 68, 64, 96);
    k_pack_wfrag<<<gs(64 * 64), 256, 0, stream>>>(vq_w1, wvq1p, 64, 64, 64);
    k_pack_wfrag<<<gs(96 * 64), 256, 0, stream>>>(cq_w0, wcq0p, 69, 64, 96);
    k_pack_wfrag<<<gs(64 * 64), 256, 0, stream>>>(cq_w1, wcq1p, 64, 64, 64);
    k_pack_wfrag<<<gs(192 * 128), 256, 0, stream>>>(cm_w0, wcm0p, 192, 128, 192);
    k_pack_wfrag<<<gs(128 * 128), 256, 0, stream>>>(cm_w1, wcm1p, 128, 128, 128);
    k_pack_wfrag<<<gs(256 * 128), 256, 0, stream>>>(ug_w0, wug0p, 256, 128, 256);
    k_pack_wfrag<<<gs(128 * 128), 256, 0, stream>>>(ug_w1, wug1p, 128, 128, 128);
    k_pack_wfrag<<<gs(128 * 64), 256, 0, stream>>>(ug_w2, wug2p, 128, 64, 128);
    k_pack_wfrag<<<gs(64 * 64), 256, 0, stream>>>(vo_w0, wvo0p, 64, 64, 64);

    // ---- 4 message-passing rounds
    for (int r = 0; r < ROUNDS_N; ++r) {
        // variables_query MLP (concat padded to K=96)
        k_assemble_vq<<<gs((size_t)V * 96), 256, 0, stream>>>(variables, v_h2, V, r);
        gemm(stream, v_h2, wvq0p, vq_b0, v_h1, V, 96, 64, 0.2f);
        gemm(stream, v_h1, wvq1p, vq_b1, vquery, V, 64, 64, 1.0f);
        // clauses_query MLP (concat padded to K=96)
        k_assemble_cq<<<gs((size_t)C * 96), 256, 0, stream>>>(clause_state, mask, S_big, C, r);
        gemm(stream, S_big, wcq0p, cq_b0, S_h1, C, 96, 64, 0.2f);
        gemm(stream, S_h1, wcq1p, cq_b1, cquery, C, 64, 64, 1.0f);
        // clause loss fwd + grads
        k_fill<<<gs((size_t)C * 64), 256, 0, stream>>>(t_cgrad, 0.0f, (size_t)C * 64);
        k_spmm_cl_soft<<<gs((size_t)E * 64), 256, 0, stream>>>(vquery, lit_idx, clause_idx, t_cgrad, E, V);
        k_cval<<<gs((size_t)C * 64), 256, 0, stream>>>(cquery, t_cgrad, cval, (size_t)C * 64);
        k_fill<<<gs((size_t)2 * V * 64), 256, 0, stream>>>(gback, 0.0f, (size_t)2 * V * 64);
        k_gback<<<gs((size_t)E * 64), 256, 0, stream>>>(cval, lit_idx, clause_idx, gback, E);
        k_vgrad<<<gs((size_t)V * 64), 256, 0, stream>>>(vquery, gback, vdw, vgrad, V);
        // clause MLP
        k_assemble_cu<<<gs((size_t)C * 192), 256, 0, stream>>>(clause_state, cval, t_cgrad, mask, S_big, C);
        gemm(stream, S_big, wcm0p, cm_b0, S_h1, C, 192, 128, 0.2f);
        gemm(stream, S_h1, wcm1p, cm_b1, S_cdata, C, 128, 128, 1.0f);
        // scatter clause_data[:, :64] to literals, scale by degree weight
        k_fill<<<gs((size_t)2 * V * 64), 256, 0, stream>>>(vl, 0.0f, (size_t)2 * V * 64);
        k_spmm_adj<<<gs((size_t)E * 64), 256, 0, stream>>>(S_cdata, lit_idx, clause_idx, vl, E);
        k_scale_rows<<<gs((size_t)2 * V * 64), 256, 0, stream>>>(vl, dw, (size_t)2 * V * 64);
        // pair-norm clause_data[:, 64:128] -> candidate clause state
        k_fill<<<gs(G_N * 64), 256, 0, stream>>>(pnsum, 0.0f, G_N * 64);
        k_fill<<<gs(G_N * 64), 256, 0, stream>>>(pnsq, 0.0f, G_N * 64);
        k_pn_sum<<<gs((size_t)C * 64), 256, 0, stream>>>(S_cdata + 64, 128, cgid, pnsum, C);
        k_pn_center<<<gs((size_t)C * 64), 256, 0, stream>>>(S_cdata + 64, 128, cgid, pnsum, inv_c, cand_c, pnsq, C);
        k_pn_finish<<<gs((size_t)C * 64), 256, 0, stream>>>(cgid, pnsq, inv_c, clause_state, cand_c, C);
        // update_gate MLP + pair-norm -> candidate variables
        k_assemble_unit<<<gs((size_t)V * 256), 256, 0, stream>>>(vgrad, variables, vl, S_big, V);
        gemm(stream, S_big, wug0p, ug_b0, v_h1, V, 256, 128, 0.2f);
        gemm(stream, v_h1, wug1p, ug_b1, v_h2, V, 128, 128, 0.2f);
        gemm(stream, v_h2, wug2p, ug_b2, nv, V, 128, 64, 1.0f);
        k_fill<<<gs(G_N * 64), 256, 0, stream>>>(pnsum, 0.0f, G_N * 64);
        k_fill<<<gs(G_N * 64), 256, 0, stream>>>(pnsq, 0.0f, G_N * 64);
        k_pn_sum<<<gs((size_t)V * 64), 256, 0, stream>>>(nv, 64, vgid, pnsum, V);
        k_pn_center<<<gs((size_t)V * 64), 256, 0, stream>>>(nv, 64, vgid, pnsum, inv_v, cand_v, pnsq, V);
        k_pn_finish<<<gs((size_t)V * 64), 256, 0, stream>>>(vgid, pnsq, inv_v, variables, cand_v, V);
        // logits
        gemm(stream, cand_v, wvo0p, vo_b0, v_hid, V, 64, 64, 0.2f);
        k_logits<<<gs(V), 256, 0, stream>>>(v_hid, vo_w1, vo_b1, logitsb, V, r);
        // evaluation: per-graph loss + SAT check
        k_fill<<<gs(C), 256, 0, stream>>>(t2, 0.0f, C);
        k_fill<<<gs(C), 256, 0, stream>>>(satv, 0.0f, C);
        k_fill<<<gs(G_N), 256, 0, stream>>>(gloss, 0.0f, G_N);
        k_set_int<<<1, 1, 0, stream>>>(satflag, 1);
        k_eval_edges<<<gs(E), 256, 0, stream>>>(logitsb, lit_idx, clause_idx, t2, satv, E, V);
        k_eval_clauses<<<gs(C), 256, 0, stream>>>(t2, satv, mask, cgid, gloss, satflag, C);
        // gated commits (use `done` from round start), then round bookkeeping
        k_commit<<<gs((size_t)V * 64), 256, 0, stream>>>(variables, cand_v, flags, (size_t)V * 64);
        k_commit<<<gs((size_t)C * 64), 256, 0, stream>>>(clause_state, cand_c, flags, (size_t)C * 64);
        k_commit<<<gs(V), 256, 0, stream>>>(last_logits, logitsb, flags, V);
        k_finish<<<1, 64, 0, stream>>>(gloss, loss_accum, flags, satflag);
    }

    k_write_out<<<gs(V + G_N + 1), 256, 0, stream>>>((float*)d_out, last_logits, loss_accum, flags, V);
}
